// rnn_70016556860048
// MI455X (gfx1250) — matrix operand order repaired and checked by reading, not compiled, not run on hardware
//
#include <hip/hip_runtime.h>
#include <hip/hip_bf16.h>

// ---------------------------------------------------------------------------
// LSTM (T=512, B=64, I=256, H=1024, O=256) on gfx1250 via bf16 WMMA.
//   Phase 1: xg = x @ W_ih^T + (b_ih+b_hh)      (parallel GEMM, WMMA)
//   Phase 2: persistent-grid sequential scan     (WMMA, W_hh in 145KB LDS,
//                                                 device barrier per step)
//   Phase 3: out = hs @ W_out^T + b_out          (parallel GEMM, WMMA)
// ---------------------------------------------------------------------------

#define T_DIM 512
#define B_DIM 64
#define I_DIM 256
#define H_DIM 1024
#define O_DIM 256
#define G4    (4 * H_DIM)       // 4096 gate columns
#define NBLK_LSTM 64            // persistent blocks; each owns 16 hidden cols
#define WPAD  8                 // bf16 pad per LDS weight row (bank spread)

typedef __bf16 bf16;
typedef bf16  v16bf __attribute__((ext_vector_type(16)));
typedef bf16  v8bf  __attribute__((ext_vector_type(8)));
typedef bf16  v4bf  __attribute__((ext_vector_type(4)));
typedef float v8f   __attribute__((ext_vector_type(8)));

// ---------------- WMMA helpers (CDNA5 16x16x32 bf16, f32 accumulate) -------

__device__ __forceinline__ v8f wmma_bf16(v16bf a, v16bf b, v8f c) {
    // (neg_a, A, neg_b, B, c_mod, C, reuse_a, reuse_b)
    return __builtin_amdgcn_wmma_f32_16x16x32_bf16(false, a, false, b,
                                                   (short)0, c, false, false);
}

__device__ __forceinline__ v16bf join16(v8bf lo, v8bf hi) {
    v16bf r;
#pragma unroll
    for (int i = 0; i < 8; ++i) { r[i] = lo[i]; r[i + 8] = hi[i]; }
    return r;
}

// A fragment: 16x32 tile of row-major [*, ld] matrix at (row0, k0).
// lane(0-15)=row M, lane-half selects contiguous K range of 16 (matches B).
__device__ __forceinline__ v16bf load_a_frag(const bf16* __restrict__ src,
                                             int ld, int row0, int k0, int lane) {
    const int m    = lane & 15;
    const int half = lane >> 4;
    const bf16* p  = src + (size_t)(row0 + m) * ld + k0 + half * 16;
    return join16(*(const v8bf*)p, *(const v8bf*)(p + 8));
}

// B fragment: 32x16 tile where B[k][n] = W[n0+n][k0+k], W row-major [*, ld].
// lane = column n, lane-half selects contiguous K range of 16.
__device__ __forceinline__ v16bf load_b_frag(const bf16* __restrict__ w,
                                             int ld, int n0, int k0, int lane) {
    const int n    = lane & 15;
    const int half = lane >> 4;
    const bf16* p  = w + (size_t)(n0 + n) * ld + k0 + half * 16;
    return join16(*(const v8bf*)p, *(const v8bf*)(p + 8));
}

// ---------------- small utility kernels ------------------------------------

__global__ void k_cvt_bf16(const float* __restrict__ in, bf16* __restrict__ out,
                           int n) {
    int i = blockIdx.x * blockDim.x + threadIdx.x;
    if (i < n) out[i] = (bf16)in[i];
}

__global__ void k_bias_sum(const float* __restrict__ a,
                           const float* __restrict__ b,
                           float* __restrict__ o, int n) {
    int i = blockIdx.x * blockDim.x + threadIdx.x;
    if (i < n) o[i] = a[i] + b[i];
}

__global__ void k_zero_u32(unsigned* __restrict__ p, int n) {
    int i = blockIdx.x * blockDim.x + threadIdx.x;
    if (i < n) p[i] = 0u;
}

// ---------------- Phase 1: xg = x @ W_ih^T + bias --------------------------
// One wave -> 16 rows x 64 cols (4 N-tiles share the A fragment). K = 256.

__global__ void __launch_bounds__(256)
k_xgemm(const bf16* __restrict__ xbf,   // [32768, 256]
        const bf16* __restrict__ wih,   // [4096, 256]
        const float* __restrict__ bias, // [4096]
        bf16* __restrict__ xg) {        // [32768, 4096]
    const int lane = threadIdx.x & 31;
    const int wid  = blockIdx.x * 8 + (threadIdx.x >> 5);
    const int rt   = wid >> 6;   // 0..2047 row tile
    const int cg   = wid & 63;   // 0..63 col group of 64

    v8f acc[4] = {v8f{}, v8f{}, v8f{}, v8f{}};
#pragma unroll
    for (int kt = 0; kt < I_DIM / 32; ++kt) {
        const int k0 = kt * 32;
        v16bf a = load_a_frag(xbf, I_DIM, rt * 16, k0, lane);
#pragma unroll
        for (int nt = 0; nt < 4; ++nt) {
            v16bf b = load_b_frag(wih, I_DIM, cg * 64 + nt * 16, k0, lane);
            acc[nt] = wmma_bf16(a, b, acc[nt]);
        }
    }
    const int n    = lane & 15;
    const int half = lane >> 4;
#pragma unroll
    for (int nt = 0; nt < 4; ++nt) {
        const int col = cg * 64 + nt * 16 + n;
        const float bs = bias[col];
#pragma unroll
        for (int r = 0; r < 8; ++r) {
            const int row = rt * 16 + half * 8 + r;
            xg[(size_t)row * G4 + col] = (bf16)(acc[nt][r] + bs);
        }
    }
}

// ---------------- device-wide barrier for the persistent scan --------------

__device__ __forceinline__ void grid_barrier(unsigned* __restrict__ bar,
                                             unsigned nblk) {
    __syncthreads();
    if (threadIdx.x == 0) {
        __threadfence();  // make this block's h stores agent-visible
        unsigned gen = __hip_atomic_load(&bar[1], __ATOMIC_RELAXED,
                                         __HIP_MEMORY_SCOPE_AGENT);
        unsigned arrived = __hip_atomic_fetch_add(&bar[0], 1u, __ATOMIC_ACQ_REL,
                                                  __HIP_MEMORY_SCOPE_AGENT);
        if (arrived == nblk - 1) {
            __hip_atomic_store(&bar[0], 0u, __ATOMIC_RELAXED,
                               __HIP_MEMORY_SCOPE_AGENT);
            __hip_atomic_fetch_add(&bar[1], 1u, __ATOMIC_RELEASE,
                                   __HIP_MEMORY_SCOPE_AGENT);
        } else {
            while (__hip_atomic_load(&bar[1], __ATOMIC_ACQUIRE,
                                     __HIP_MEMORY_SCOPE_AGENT) == gen) {
                __builtin_amdgcn_s_sleep(2);
            }
        }
    }
    __syncthreads();
}

// ---------------- Phase 2: persistent LSTM scan ----------------------------
// Block g owns hidden columns [16g, 16g+16).  Its 128 KB W_hh slice (4 gates
// x 16 cols x 1024 K, bf16) is staged once into LDS (320 KB/WGP on CDNA5)
// and served from ds_load for all 512 steps; h comes from L2 each step; the
// cell state c lives in registers for the whole scan.

__global__ void __launch_bounds__(256)
k_lstm(const bf16* __restrict__ xg,   // [T*B, 4096]
       const bf16* __restrict__ whh,  // [4096, 1024]
       bf16* __restrict__ hbuf,       // [2][64*1024] double buffer
       bf16* __restrict__ hs,         // [T*B, 1024]
       unsigned* __restrict__ bar) {
    const int g    = blockIdx.x;          // hidden slice id (0..63)
    const int tid  = threadIdx.x;
    const int lane = tid & 31;
    const int wave = tid >> 5;            // 0..7
    const int bt   = wave & 3;            // batch tile (16 rows)
    const int qp   = wave >> 2;           // gate pair: {0,1} or {2,3}
    const int q0   = qp * 2, q1 = qp * 2 + 1;

    // Padded row stride => B-fragment lanes land on distinct LDS bank groups.
    __shared__ bf16  wlds[4][16][H_DIM + WPAD];   // 132,096 B
    __shared__ float gbuf[4][B_DIM][16];          //  16,384 B

    // Stage this block's W_hh slice into LDS (16-byte chunks, coalesced).
    for (int idx = tid; idx < 64 * (H_DIM / 8); idx += 256) {
        const int c  = idx >> 7;            // 0..63 local column
        const int kk = (idx & 127) * 8;     // K offset
        const int q  = c >> 4, j = c & 15;
        const int gcol = q * H_DIM + g * 16 + j;
        *(v8bf*)&wlds[q][j][kk] = *(const v8bf*)(whh + (size_t)gcol * H_DIM + kk);
    }

    // Zero h0 in buffer 0 (ws is poisoned; buffer 1 is written before read).
    const int eb = tid >> 2;              // batch row this thread owns
    const int j0 = (tid & 3) * 4;         // 4 consecutive hidden cols
    {
        v4bf z = {};
        *(v4bf*)(hbuf + (size_t)eb * H_DIM + g * 16 + j0) = z;
    }
    float creg[4] = {0.f, 0.f, 0.f, 0.f}; // cell state lives here for all T

    grid_barrier(bar, NBLK_LSTM);         // h0 visible + LDS staged

    const int n    = lane & 15;
    const int half = lane >> 4;
    const bf16* bp0 = &wlds[q0][n][half * 16];  // B-frag base, gate q0
    const bf16* bp1 = &wlds[q1][n][half * 16];  // B-frag base, gate q1
    const int ncol0 = q0 * H_DIM + g * 16;
    const int ncol1 = q1 * H_DIM + g * 16;

    for (int t = 0; t < T_DIM; ++t) {
        const bf16* hin  = hbuf + (size_t)(t & 1) * (B_DIM * H_DIM);
        bf16*       hout = hbuf + (size_t)((t + 1) & 1) * (B_DIM * H_DIM);

        v8f acc0 = {}, acc1 = {};
#pragma unroll 4
        for (int kt = 0; kt < H_DIM / 32; ++kt) {
            const int k0 = kt * 32;
            v16bf a  = load_a_frag(hin, H_DIM, bt * 16, k0, lane);
            v16bf b0 = join16(*(const v8bf*)(bp0 + k0),
                              *(const v8bf*)(bp0 + k0 + 8));
            v16bf b1 = join16(*(const v8bf*)(bp1 + k0),
                              *(const v8bf*)(bp1 + k0 + 8));
            acc0 = wmma_bf16(a, b0, acc0);
            acc1 = wmma_bf16(a, b1, acc1);
        }

        // acc + xg  -> LDS gate buffer (layout [gate][batch][hidden-in-slice])
        const bf16* xgrow = xg + (size_t)t * B_DIM * G4;
#pragma unroll
        for (int r = 0; r < 8; ++r) {
            const int b = bt * 16 + half * 8 + r;
            gbuf[q0][b][n] = acc0[r] + (float)xgrow[(size_t)b * G4 + ncol0 + n];
            gbuf[q1][b][n] = acc1[r] + (float)xgrow[(size_t)b * G4 + ncol1 + n];
        }
        __syncthreads();

        // Activations + cell update for the 4 (b, j) elements this thread owns.
        {
            v4bf hv;
#pragma unroll
            for (int jj = 0; jj < 4; ++jj) {
                const int j = j0 + jj;
                const float ig = 1.f / (1.f + __expf(-gbuf[0][eb][j]));
                const float fg = 1.f / (1.f + __expf(-gbuf[1][eb][j]));
                const float gg = tanhf(gbuf[2][eb][j]);
                const float og = 1.f / (1.f + __expf(-gbuf[3][eb][j]));
                creg[jj] = fg * creg[jj] + ig * gg;
                hv[jj]   = (bf16)(og * tanhf(creg[jj]));
            }
            *(v4bf*)(hout + (size_t)eb * H_DIM + g * 16 + j0) = hv;
            *(v4bf*)(hs + ((size_t)t * B_DIM + eb) * H_DIM + g * 16 + j0) = hv;
        }
        grid_barrier(bar, NBLK_LSTM);  // release h_t, protect gbuf reuse
    }
}

// ---------------- Phase 3: out = hs @ W_out^T + b_out ----------------------

__global__ void __launch_bounds__(256)
k_outgemm(const bf16* __restrict__ hsb,   // [32768, 1024]
          const bf16* __restrict__ wout,  // [256, 1024]
          const float* __restrict__ bout, // [256]
          float* __restrict__ out) {      // [32768, 256]
    const int lane = threadIdx.x & 31;
    const int wid  = blockIdx.x * 8 + (threadIdx.x >> 5);
    const int rt   = wid >> 4;   // 0..2047
    const int ct   = wid & 15;   // 0..15

    v8f acc = {};
#pragma unroll 4
    for (int kt = 0; kt < H_DIM / 32; ++kt) {
        const int k0 = kt * 32;
        v16bf a = load_a_frag(hsb, H_DIM, rt * 16, k0, lane);
        v16bf b = load_b_frag(wout, H_DIM, ct * 16, k0, lane);
        acc = wmma_bf16(a, b, acc);
    }
    const int n    = lane & 15;
    const int half = lane >> 4;
    const int col  = ct * 16 + n;
    const float bs = bout[col];
#pragma unroll
    for (int r = 0; r < 8; ++r) {
        const int row = rt * 16 + half * 8 + r;
        out[(size_t)row * O_DIM + col] = acc[r] + bs;
    }
}

// ---------------- host-side launch -----------------------------------------

extern "C" void kernel_launch(void* const* d_in, const int* in_sizes, int n_in,
                              void* d_out, int out_size, void* d_ws,
                              size_t ws_size, hipStream_t stream) {
    (void)in_sizes; (void)n_in; (void)out_size; (void)ws_size;
    const float* x     = (const float*)d_in[0];  // [512,64,256]
    const float* W_ih  = (const float*)d_in[1];  // [4096,256]
    const float* W_hh  = (const float*)d_in[2];  // [4096,1024]
    const float* b_ih  = (const float*)d_in[3];  // [4096]
    const float* b_hh  = (const float*)d_in[4];  // [4096]
    const float* W_out = (const float*)d_in[5];  // [256,1024]
    const float* b_out = (const float*)d_in[6];  // [256]
    float* out = (float*)d_out;                  // [512,64,256]

    const size_t NX = (size_t)T_DIM * B_DIM * I_DIM;   // 8,388,608
    const size_t NWI = (size_t)G4 * I_DIM;             // 1,048,576
    const size_t NWH = (size_t)G4 * H_DIM;             // 4,194,304
    const size_t NWO = (size_t)O_DIM * H_DIM;          // 262,144
    const size_t NROWS = (size_t)T_DIM * B_DIM;        // 32,768

    char* ws = (char*)d_ws;
    size_t off = 0;
    auto alloc = [&](size_t bytes) {
        char* p = ws + off;
        off = (off + bytes + 255) & ~(size_t)255;
        return p;
    };
    bf16*     x_bf   = (bf16*)alloc(NX * 2);
    bf16*     wih_bf = (bf16*)alloc(NWI * 2);
    bf16*     whh_bf = (bf16*)alloc(NWH * 2);
    bf16*     wout_bf= (bf16*)alloc(NWO * 2);
    float*    bias   = (float*)alloc(G4 * 4);
    bf16*     xg     = (bf16*)alloc(NROWS * G4 * 2);     // 256 MB
    bf16*     hbuf   = (bf16*)alloc((size_t)2 * B_DIM * H_DIM * 2);
    bf16*     hs     = (bf16*)alloc(NROWS * H_DIM * 2);  // 64 MB
    unsigned* bar    = (unsigned*)alloc(2 * 4);

    // --- convert inputs to bf16, build combined bias, reset barrier ---
    k_cvt_bf16<<<(int)((NX + 255) / 256), 256, 0, stream>>>(x, x_bf, (int)NX);
    k_cvt_bf16<<<(int)((NWI + 255) / 256), 256, 0, stream>>>(W_ih, wih_bf, (int)NWI);
    k_cvt_bf16<<<(int)((NWH + 255) / 256), 256, 0, stream>>>(W_hh, whh_bf, (int)NWH);
    k_cvt_bf16<<<(int)((NWO + 255) / 256), 256, 0, stream>>>(W_out, wout_bf, (int)NWO);
    k_bias_sum<<<(G4 + 255) / 256, 256, 0, stream>>>(b_ih, b_hh, bias, G4);
    k_zero_u32<<<1, 32, 0, stream>>>(bar, 2);

    // --- phase 1: xg GEMM (2048 row-tiles x 64 col-groups, 8 waves/block) ---
    k_xgemm<<<16384, 256, 0, stream>>>(x_bf, wih_bf, bias, xg);

    // --- phase 2: persistent sequential scan over T (W_hh slice in LDS) ---
    k_lstm<<<NBLK_LSTM, 256, 0, stream>>>(xg, whh_bf, hbuf, hs, bar);

    // --- phase 3: output projection ---
    k_outgemm<<<4096, 256, 0, stream>>>(hs, wout_bf, b_out, out);
}
